// Model_455266534068
// MI455X (gfx1250) — matrix operand order repaired and checked by reading, not compiled, not run on hardware
//
#include <hip/hip_runtime.h>
#include <hip/hip_bf16.h>
#include <stdint.h>

// ---------------------------------------------------------------------------
// 2-layer LSTM LM forward on MI455X (gfx1250), bf16 WMMA everywhere.
//   HID=1024, 4H=4096, T=64, B=32, T*B=2048, VOCAB=32000
// Strategy:
//   * x-path GEMMs batched over all timesteps (M=2048) -> big WMMA GEMM
//   * per-step h-path GEMM (M=32) fused with gate math in one kernel,
//     Wh stays L2-resident (8.4 MB bf16 << 192 MB L2)
//   * FC projection (2048x32000x1024) -> 64x64 wave-tile WMMA GEMM with
//     nontemporal logits stores (262 MB streamed once; protect L2-resident
//     fc_W) straight into d_out
//   * All inner GEMM loops strength-reduced: loads use base-pointer +
//     immediate offsets only, so no VALU address math (and no WMMA->VALU
//     hazard NOPs) inside the hot loop.
// ---------------------------------------------------------------------------

typedef unsigned short u16;
typedef __attribute__((ext_vector_type(16))) __bf16 v16bf;
typedef __attribute__((ext_vector_type(8)))  float  v8f;

union Frag32 { v16bf v; uint4 q[2]; };

__device__ __forceinline__ u16 f32_to_bf16(float f) {
  union { float f; unsigned int u; } x; x.f = f;
  unsigned int r = 0x7FFFu + ((x.u >> 16) & 1u);   // round-to-nearest-even
  return (u16)((x.u + r) >> 16);
}

__device__ __forceinline__ float fsigmoid(float x) {
  return 1.0f / (1.0f + __expf(-x));
}
__device__ __forceinline__ float ftanh_fast(float x) {
  float t = __expf(-2.0f * x);
  return (1.0f - t) / (1.0f + t);
}

__device__ __forceinline__ v8f wmma_bf16(const Frag32& a, const Frag32& b,
                                         v8f c) {
  return __builtin_amdgcn_wmma_f32_16x16x32_bf16(
      false, a.v, false, b.v, (short)0, c, false, false);
}

// ---------------------------------------------------------------------------
// Generic GEMM: C[M,N] = A[M,K](bf16) * Bt[N,K](bf16)^T + bias[N], fp32 out.
// 128 threads = 4 waves; wave tile 64Mx64N (4 A-frags x 4 B-frags, 16 WMMA
// per K-step, 16 b128 loads -> ~33 FLOP/loaded-byte).
// Requires M%64==0, N%256==0, K%128==0 (true for all our shapes).
// NT_STORE=true streams C past L2 (logits); false keeps RT (gx buffers).
// ---------------------------------------------------------------------------
template <bool NT_STORE>
__global__ __launch_bounds__(128) void gemm_bf16_bias(
    const u16* __restrict__ A, const u16* __restrict__ Bt,
    const float* __restrict__ bias, float* __restrict__ C,
    int M, int N, int K)
{
  const int lane = threadIdx.x & 31;
  const int wave = threadIdx.x >> 5;
  const int col  = lane & 15;
  const int half = lane >> 4;
  const int n0 = blockIdx.x * 256 + wave * 64;
  const int m0 = blockIdx.y * 64;

  // per-row pointers; advanced by 32 elements (64B) per k-step so the inner
  // loop is pure loads (immediate offsets) + WMMA.
  const u16* ap[4];
  const u16* bp[4];
#pragma unroll
  for (int mt = 0; mt < 4; ++mt)
    ap[mt] = A + (size_t)(m0 + mt * 16 + col) * K + half * 8;
#pragma unroll
  for (int nt = 0; nt < 4; ++nt)
    bp[nt] = Bt + (size_t)(n0 + nt * 16 + col) * K + half * 8;

  v8f acc[4][4] = {};

  for (int kc = 0; kc < K; kc += 128) {
    // cover L2->WGP latency on the streaming weight rows (global_prefetch_b8)
#pragma unroll
    for (int nt = 0; nt < 4; ++nt)
      __builtin_prefetch(bp[nt] + 128, 0, 1);
#pragma unroll
    for (int ku = 0; ku < 4; ++ku) {
      Frag32 a[4];
#pragma unroll
      for (int mt = 0; mt < 4; ++mt) {
        a[mt].q[0] = *(const uint4*)(ap[mt]);       // K = k+half*8   .. +7
        a[mt].q[1] = *(const uint4*)(ap[mt] + 16);  // K = k+16+half*8.. +7
        ap[mt] += 32;
      }
#pragma unroll
      for (int nt = 0; nt < 4; ++nt) {
        Frag32 b;
        b.q[0] = *(const uint4*)(bp[nt]);           // K = k+half*8   .. +7
        b.q[1] = *(const uint4*)(bp[nt] + 16);      // K = k+16+half*8.. +7
        bp[nt] += 32;
#pragma unroll
        for (int mt = 0; mt < 4; ++mt)
          acc[mt][nt] = wmma_bf16(a[mt], b, acc[mt][nt]);
      }
    }
  }

#pragma unroll
  for (int mt = 0; mt < 4; ++mt)
#pragma unroll
    for (int nt = 0; nt < 4; ++nt) {
      const int n  = n0 + nt * 16 + col;
      const float bv = bias[n];
#pragma unroll
      for (int r = 0; r < 8; ++r) {
        const int m = m0 + mt * 16 + r + half * 8;
        const float v = acc[mt][nt][r] + bv;
        if (NT_STORE)
          __builtin_nontemporal_store(v, &C[(size_t)m * N + n]);
        else
          C[(size_t)m * N + n] = v;
      }
    }
}

// ---------------------------------------------------------------------------
// One LSTM timestep: gh = Hprev(bf16,[32x1024]) @ Wh^T; fuse with gx, biases,
// gates, c/h update. 128 threads = 4 waves, one gate per wave; block covers
// 64 hidden units. Grid = 16 blocks. Wh row-major [4096x1024] bf16.
// K=1024 is a constant here, so mt/nt tile offsets are compile-time byte
// immediates (16 rows * 1024 * 2B = 32768B, fits the 24-bit IOFFSET) and the
// whole k-loop runs off two loop-carried pointers.
// ---------------------------------------------------------------------------
__global__ __launch_bounds__(128) void lstm_step(
    const u16* __restrict__ Hprev,   // [32 x 1024] bf16 (h at t-1)
    const u16* __restrict__ Whb,     // [4096 x 1024] bf16
    const float* __restrict__ bh,    // [4096]
    const float* __restrict__ gx,    // [32 x 4096] fp32 (= x@Wx^T + bx, this t)
    const float* c_in,               // [32 x 1024] (may alias c_out)
    float* c_out,                    // [32 x 1024]
    float* __restrict__ h_out,       // [32 x 1024] fp32
    u16* __restrict__ y_out)         // [32 x 1024] bf16 (this t's slice)
{
  __shared__ float gsm[4][32][64];   // 32 KB: [gate][row][hidden-unit]
  const int lane = threadIdx.x & 31;
  const int g    = threadIdx.x >> 5;     // gate 0..3 (i,f,o,n)
  const int col  = lane & 15;
  const int half = lane >> 4;
  const int j0   = blockIdx.x * 64;      // hidden-unit block

  const u16* ap = Hprev + (size_t)col * 1024 + half * 8;
  const u16* bp = Whb + (size_t)(g * 1024 + j0 + col) * 1024 + half * 8;

  v8f acc[2][4] = {};

  for (int kc = 0; kc < 1024; kc += 128) {
#pragma unroll
    for (int nt = 0; nt < 4; ++nt)
      __builtin_prefetch(bp + nt * 16384 + 128, 0, 1);
#pragma unroll
    for (int ku = 0; ku < 4; ++ku) {
      Frag32 a[2];
      a[0].q[0] = *(const uint4*)(ap);            // row col,     K chunk lo
      a[0].q[1] = *(const uint4*)(ap + 16);       // row col,     K chunk hi
      a[1].q[0] = *(const uint4*)(ap + 16384);    // row col+16 (32768B imm)
      a[1].q[1] = *(const uint4*)(ap + 16400);
#pragma unroll
      for (int nt = 0; nt < 4; ++nt) {
        Frag32 b;
        const u16* bq = bp + nt * 16384;          // +nt*32768B immediates
        b.q[0] = *(const uint4*)(bq);
        b.q[1] = *(const uint4*)(bq + 16);
        acc[0][nt] = wmma_bf16(a[0], b, acc[0][nt]);
        acc[1][nt] = wmma_bf16(a[1], b, acc[1][nt]);
      }
      ap += 32;
      bp += 32;
    }
  }

  // gh + gx + bh -> LDS, laid out for the gate combine
#pragma unroll
  for (int mt = 0; mt < 2; ++mt)
#pragma unroll
    for (int nt = 0; nt < 4; ++nt) {
      const int hu = nt * 16 + col;
      const int n  = g * 1024 + j0 + hu;
      const float bv = bh[n];
#pragma unroll
      for (int r = 0; r < 8; ++r) {
        const int m = mt * 16 + r + half * 8;
        gsm[g][m][hu] = acc[mt][nt][r] + gx[m * 4096 + n] + bv;
      }
    }
  __syncthreads();

  // elementwise cell update: 2048 cells / 128 threads = 16 each
#pragma unroll 4
  for (int it = 0; it < 16; ++it) {
    const int cell = it * 128 + threadIdx.x;
    const int m  = cell >> 6;
    const int hu = cell & 63;
    const float i = fsigmoid(gsm[0][m][hu]);
    const float f = fsigmoid(gsm[1][m][hu]);
    const float o = fsigmoid(gsm[2][m][hu]);
    const float n = ftanh_fast(gsm[3][m][hu]);
    const int idx = m * 1024 + j0 + hu;
    const float c = f * c_in[idx] + i * n;
    c_out[idx] = c;
    const float h = o * ftanh_fast(c);
    h_out[idx] = h;
    y_out[idx] = f32_to_bf16(h);
  }
}

// ---------------------------------------------------------------------------
// Helpers: fp32->bf16 convert, embedding gather, final-state copy
// ---------------------------------------------------------------------------
__global__ void convert_f32_bf16(const float* __restrict__ in,
                                 u16* __restrict__ out, long n) {
  long i = (long)blockIdx.x * blockDim.x + threadIdx.x;
  const long stride = (long)gridDim.x * blockDim.x;
  for (; i < n; i += stride) out[i] = f32_to_bf16(in[i]);
}

__global__ void embed_bf16(const int* __restrict__ x,
                           const float* __restrict__ emb,
                           u16* __restrict__ e) {
  const int row = blockIdx.x;                 // 0..2047 = t*32+b
  const int tok = x[row];
  const float* src = emb + (size_t)tok * 1024;
  u16* dst = e + (size_t)row * 1024;
  for (int k = threadIdx.x; k < 1024; k += blockDim.x)
    dst[k] = f32_to_bf16(src[k]);
}

__global__ void finalize_states(const float* __restrict__ h0,
                                const float* __restrict__ c0,
                                const float* __restrict__ h1,
                                const float* __restrict__ c1,
                                float* __restrict__ out) {
  const int i = blockIdx.x * blockDim.x + threadIdx.x;  // 0..131071
  const int N = 32 * 1024;
  if (i < N)            out[i] = h0[i];
  else if (i < 2 * N)   out[i] = c0[i - N];
  else if (i < 3 * N)   out[i] = h1[i - 2 * N];
  else if (i < 4 * N)   out[i] = c1[i - 3 * N];
}

// ---------------------------------------------------------------------------
extern "C" void kernel_launch(void* const* d_in, const int* in_sizes, int n_in,
                              void* d_out, int out_size, void* d_ws, size_t ws_size,
                              hipStream_t stream) {
  (void)in_sizes; (void)n_in; (void)out_size; (void)ws_size;
  const int   H = 1024, FH = 4096, TB = 2048, V = 32000, T = 64, B = 32;

  const int*   x    = (const int*)  d_in[0];
  const float* embW = (const float*)d_in[1];
  const float* Wx0  = (const float*)d_in[2];
  const float* Wh0  = (const float*)d_in[3];
  const float* bx0  = (const float*)d_in[4];
  const float* bh0  = (const float*)d_in[5];
  const float* Wx1  = (const float*)d_in[6];
  const float* Wh1  = (const float*)d_in[7];
  const float* bx1  = (const float*)d_in[8];
  const float* bh1  = (const float*)d_in[9];
  const float* fcW  = (const float*)d_in[10];
  const float* fcb  = (const float*)d_in[11];
  const float* h0   = (const float*)d_in[12];
  const float* c0   = (const float*)d_in[13];
  const float* h1   = (const float*)d_in[14];
  const float* c1   = (const float*)d_in[15];
  float* out = (float*)d_out;

  // workspace carve-up (256B aligned)
  char* ws = (char*)d_ws;
  size_t off = 0;
  auto alloc = [&](size_t bytes) -> char* {
    char* p = ws + off;
    off += (bytes + 255) & ~(size_t)255;
    return p;
  };
  u16*   eB    = (u16*)alloc((size_t)TB * H * 2);     // embedded seq, bf16
  u16*   Wx0B  = (u16*)alloc((size_t)FH * H * 2);
  u16*   Wh0B  = (u16*)alloc((size_t)FH * H * 2);
  u16*   Wx1B  = (u16*)alloc((size_t)FH * H * 2);
  u16*   Wh1B  = (u16*)alloc((size_t)FH * H * 2);
  u16*   fcWB  = (u16*)alloc((size_t)V  * H * 2);
  float* gx    = (float*)alloc((size_t)TB * FH * 4);  // reused for both layers
  u16*   y0B   = (u16*)alloc((size_t)TB * H * 2);     // layer0 h seq, bf16
  u16*   y1B   = (u16*)alloc((size_t)TB * H * 2);     // layer1 h seq, bf16
  u16*   h0iB  = (u16*)alloc((size_t)B * H * 2);      // bf16(h0)
  u16*   h1iB  = (u16*)alloc((size_t)B * H * 2);      // bf16(h1)
  float* c0buf = (float*)alloc((size_t)B * H * 4);
  float* c1buf = (float*)alloc((size_t)B * H * 4);
  float* h0buf = (float*)alloc((size_t)B * H * 4);
  float* h1buf = (float*)alloc((size_t)B * H * 4);

  // 1) weights & initial h-states to bf16
  convert_f32_bf16<<<2048, 256, 0, stream>>>(Wx0, Wx0B, (long)FH * H);
  convert_f32_bf16<<<2048, 256, 0, stream>>>(Wh0, Wh0B, (long)FH * H);
  convert_f32_bf16<<<2048, 256, 0, stream>>>(Wx1, Wx1B, (long)FH * H);
  convert_f32_bf16<<<2048, 256, 0, stream>>>(Wh1, Wh1B, (long)FH * H);
  convert_f32_bf16<<<8192, 256, 0, stream>>>(fcW, fcWB, (long)V * H);
  convert_f32_bf16<<<128, 256, 0, stream>>>(h0, h0iB, (long)B * H);
  convert_f32_bf16<<<128, 256, 0, stream>>>(h1, h1iB, (long)B * H);

  // 2) embedding gather -> bf16
  embed_bf16<<<TB, 256, 0, stream>>>(x, embW, eB);

  // 3) layer 0: batched x-path GEMM, then 64 sequential fused steps
  gemm_bf16_bias<false><<<dim3(FH / 256, TB / 64), 128, 0, stream>>>(
      eB, Wx0B, bx0, gx, TB, FH, H);
  for (int t = 0; t < T; ++t) {
    const u16* Hprev = (t == 0) ? h0iB : (y0B + (size_t)(t - 1) * B * H);
    const float* cin = (t == 0) ? c0 : c0buf;
    lstm_step<<<H / 64, 128, 0, stream>>>(
        Hprev, Wh0B, bh0, gx + (size_t)t * B * FH,
        cin, c0buf, h0buf, y0B + (size_t)t * B * H);
  }

  // 4) layer 1
  gemm_bf16_bias<false><<<dim3(FH / 256, TB / 64), 128, 0, stream>>>(
      y0B, Wx1B, bx1, gx, TB, FH, H);
  for (int t = 0; t < T; ++t) {
    const u16* Hprev = (t == 0) ? h1iB : (y1B + (size_t)(t - 1) * B * H);
    const float* cin = (t == 0) ? c1 : c1buf;
    lstm_step<<<H / 64, 128, 0, stream>>>(
        Hprev, Wh1B, bh1, gx + (size_t)t * B * FH,
        cin, c1buf, h1buf, y1B + (size_t)t * B * H);
  }

  // 5) FC projection -> logits (streaming NT stores into d_out)
  gemm_bf16_bias<true><<<dim3(V / 256, TB / 64), 128, 0, stream>>>(
      y1B, fcWB, fcb, out, TB, V, H);

  // 6) final h/c states appended after logits
  finalize_states<<<512, 256, 0, stream>>>(
      h0buf, c0buf, h1buf, c1buf, out + (size_t)TB * V);
}
